// MultiHeadAttention_9380208574843
// MI455X (gfx1250) — matrix operand order repaired and checked by reading, not compiled, not run on hardware
//
#include <hip/hip_runtime.h>
#include <hip/hip_bf16.h>

#define NH   16
#define HS   64
#define EMB  1024
#define TT   2048
#define NB   4
#define BT   (NB * TT)   // 8192

// LDS row stride for staged weight panels: 2048B data + 2x16B pad (TDM pad:
// 4 dwords per 256 dwords) -> 2080B = 520 dwords; every 16B chunk aligned.
#define LDS_ROW 2080

typedef _Float16 half_t;
typedef __attribute__((ext_vector_type(16))) _Float16 v16h;
typedef __attribute__((ext_vector_type(8)))  _Float16 v8h;
typedef __attribute__((ext_vector_type(8)))  float    v8f;

typedef unsigned int u32;
typedef __attribute__((ext_vector_type(4))) u32 v4u32;
typedef __attribute__((ext_vector_type(8))) int v8i32;
typedef __attribute__((ext_vector_type(4))) int v4i32;

// ---------------------------------------------------------------------------
// WMMA fragment helpers (wave32, V_WMMA_F32_16X16X32_F16; contiguous k = 16*(lane/16) + s in A and B)
// ---------------------------------------------------------------------------
__device__ __forceinline__ v16h load_a_frag(const half_t* base, int ld) {
  const int lane = threadIdx.x & 31;
  const int m = lane & 15, hi = lane >> 4;
  const half_t* p = base + m * ld + hi * 16;
  v8h lo = *(const v8h*)(p);
  v8h hh = *(const v8h*)(p + 8);
  v16h r;
#pragma unroll
  for (int e = 0; e < 8; ++e) { r[e] = lo[e]; r[8 + e] = hh[e]; }
  return r;
}

__device__ __forceinline__ v16h load_b_frag(const half_t* base, int ld) {
  const int lane = threadIdx.x & 31;
  const int n = lane & 15, hi = lane >> 4;
  return *(const v16h*)(base + n * ld + hi * 16);
}

// B-fragment from a TDM-staged LDS panel (row stride LDS_ROW bytes, +16B pad
// after each 1024B of data within a row).
__device__ __forceinline__ v16h load_b_lds(const half_t* lds, int nbase, int kbase) {
  const int lane = threadIdx.x & 31;
  const int n = nbase + (lane & 15);
  const int kb = (kbase + (lane >> 4) * 16) * 2;        // byte offset in row data
  const char* p = (const char*)lds + n * LDS_ROW + kb + ((kb >> 10) << 4);
  v8h lo = *(const v8h*)(p);
  v8h hh = *(const v8h*)(p + 16);
  v16h r;
#pragma unroll
  for (int e = 0; e < 8; ++e) { r[e] = lo[e]; r[8 + e] = hh[e]; }
  return r;
}

__device__ __forceinline__ v8f wmma_f16(v16h a, v16h b, v8f c) {
  return __builtin_amdgcn_wmma_f32_16x16x32_f16(false, a, false, b,
                                                (short)0, c, false, false);
}

// ---------------------------------------------------------------------------
// Tensor Data Mover: async 2D tile load Global -> LDS (ISA ch.8 D# layout).
// data_size=2B, tile = tile_d0 x tile_d1, pad 4 dwords per 256 dwords.
// ---------------------------------------------------------------------------
__device__ __forceinline__ void tdm_load_panel(const void* gsrc, u32 lds_byte,
                                               u32 tensor_d0, u32 tensor_d1,
                                               u32 tile_d0, u32 tile_d1,
                                               u32 stride_d0) {
  const unsigned long long ga = (unsigned long long)(uintptr_t)gsrc;
  v4u32 g0;
  g0[0] = 1u;                                   // count=1, user-mode load
  g0[1] = lds_byte;                             // lds_addr (bytes)
  g0[2] = (u32)(ga & 0xffffffffu);              // global_addr[31:0]
  g0[3] = (u32)((ga >> 32) & 0x1ffffffu) | (2u << 30);  // addr[56:32] | type=2
  v8i32 g1;
  // data_size=1(2B) | pad_enable | pad_interval=7(256dw) | pad_amount=3(4dw)
  g1[0] = (int)((1u << 16) | (1u << 20) | (7u << 22) | (3u << 25));
  g1[1] = (int)((tensor_d0 & 0xffffu) << 16);
  g1[2] = (int)((tensor_d0 >> 16) | ((tensor_d1 & 0xffffu) << 16));
  g1[3] = (int)((tensor_d1 >> 16) | (tile_d0 << 16));
  g1[4] = (int)(tile_d1 & 0xffffu);             // tile_dim1 ; tile_dim2=0
  g1[5] = (int)stride_d0;                       // tensor_dim0_stride[31:0]
  g1[6] = 0;
  g1[7] = 0;
  v4i32 z4 = {0, 0, 0, 0};
#if __clang_major__ >= 23
  v8i32 z8 = {0, 0, 0, 0, 0, 0, 0, 0};
  __builtin_amdgcn_tensor_load_to_lds(g0, g1, z4, z4, z8, 0);
#else
  __builtin_amdgcn_tensor_load_to_lds(g0, g1, z4, z4, 0);
#endif
}

// Low 32 bits of the flat address = LDS byte offset (ISA 10.2 aperture rules).
__device__ __forceinline__ u32 lds_addr_of(const void* p) {
  return (u32)(uintptr_t)p;
}

// ---------------------------------------------------------------------------
// Preparation kernels: fp32 -> f16, weight transposes
// ---------------------------------------------------------------------------
__global__ void k_cvt_x(const float* __restrict__ x, half_t* __restrict__ xh, int n) {
  int i = blockIdx.x * blockDim.x + threadIdx.x;
  const int stride = gridDim.x * blockDim.x;
  for (; i < n; i += stride) xh[i] = (half_t)x[i];
}

// W [H, E, HS] (f32) -> Wt [H, HS, E] (f16)
__global__ void k_cvt_w_qkv(const float* __restrict__ W, half_t* __restrict__ Wt) {
  int i = blockIdx.x * blockDim.x + threadIdx.x;
  if (i >= NH * EMB * HS) return;
  const int h = i / (EMB * HS);
  const int rem = i % (EMB * HS);
  const int e = rem / HS, d = rem % HS;
  Wt[(h * HS + d) * EMB + e] = (half_t)W[i];
}

// Wo [K=E, N=E] (f32) -> WoT [N, K] (f16)
__global__ void k_cvt_wo(const float* __restrict__ Wo, half_t* __restrict__ WoT) {
  int i = blockIdx.x * blockDim.x + threadIdx.x;
  if (i >= EMB * EMB) return;
  const int k = i / EMB, n = i % EMB;
  WoT[n * EMB + k] = (half_t)Wo[i];
}

// ---------------------------------------------------------------------------
// Shared GEMM body: C[32x64] per wave (register-blocked, 2 M-subtiles so each
// B-fragment feeds two WMMAs), B panel [64 x 1024] staged in LDS by the TDM,
// A rows streamed from global with register double-buffering.
// ---------------------------------------------------------------------------
__device__ __forceinline__ void gemm_panel2(const half_t* __restrict__ arow,
                                            const half_t* __restrict__ ldsW,
                                            v8f acc0[4], v8f acc1[4]) {
  v16h a0 = load_a_frag(arow, EMB);
  v16h a1 = load_a_frag(arow + 16 * EMB, EMB);
  for (int k0 = 0; k0 < EMB; k0 += 32) {
    v16h a0n = a0, a1n = a1;
    if (k0 + 32 < EMB) {                       // prefetch next K-step
      a0n = load_a_frag(arow + k0 + 32, EMB);
      a1n = load_a_frag(arow + 16 * EMB + k0 + 32, EMB);
    }
#pragma unroll
    for (int nt = 0; nt < 4; ++nt) {
      v16h b = load_b_lds(ldsW, nt * 16, k0);  // shared by both M-subtiles
      acc0[nt] = wmma_f16(a0, b, acc0[nt]);
      acc1[nt] = wmma_f16(a1, b, acc1[nt]);
    }
    a0 = a0n; a1 = a1n;
  }
}

// ---------------------------------------------------------------------------
// QKV projection: [BT, E] x [E, HS] per head. grid=(BT/256, NH, 3), 256 thr.
// Weight panel (64 x 1024 f16, 130KB padded) DMA'd to LDS once per workgroup.
// Q,K stored [b,h][T,HS]; V stored transposed [b,h][HS,T].
// ---------------------------------------------------------------------------
__global__ void __launch_bounds__(256)
k_qkv(const half_t* __restrict__ xh,
      const half_t* __restrict__ Wtq, const half_t* __restrict__ Wtk,
      const half_t* __restrict__ Wtv,
      half_t* __restrict__ Q, half_t* __restrict__ K, half_t* __restrict__ Vt) {
  __shared__ __align__(32) half_t ldsW[64 * LDS_ROW / 2];   // 130 KB
  const int wave = threadIdx.x >> 5;
  const int lane = threadIdx.x & 31;
  const int m0 = blockIdx.x * 256 + wave * 32;   // global row in [0, BT)
  const int h = blockIdx.y;
  const int which = blockIdx.z;                  // 0=Q 1=K 2=V
  const half_t* W =
      (which == 0 ? Wtq : (which == 1 ? Wtk : Wtv)) + (size_t)h * HS * EMB;

  if (wave == 0) {                               // one TDM per workgroup
    tdm_load_panel(W, lds_addr_of(ldsW), EMB, HS, EMB, HS, EMB);
    __builtin_amdgcn_s_wait_tensorcnt(0);
  }
  __syncthreads();

  v8f acc0[4] = {}, acc1[4] = {};
  gemm_panel2(xh + (size_t)m0 * EMB, ldsW, acc0, acc1);

  const int bidx = m0 / TT, t0 = m0 % TT;
  const int hi = lane >> 4, nlo = lane & 15;
  const int bh = bidx * NH + h;
#pragma unroll
  for (int mt = 0; mt < 2; ++mt) {
    const v8f* acc = mt ? acc1 : acc0;
    const int tb = t0 + mt * 16;
    if (which == 2) {
      half_t* __restrict__ vt = Vt + (size_t)bh * HS * TT;
#pragma unroll
      for (int nt = 0; nt < 4; ++nt)
#pragma unroll
        for (int r = 0; r < 8; ++r)
          vt[(nt * 16 + nlo) * TT + tb + hi * 8 + r] = (half_t)acc[nt][r];
    } else {
      half_t* __restrict__ o = (which == 0 ? Q : K) + (size_t)bh * TT * HS;
#pragma unroll
      for (int nt = 0; nt < 4; ++nt)
#pragma unroll
        for (int r = 0; r < 8; ++r)
          o[(tb + hi * 8 + r) * HS + nt * 16 + nlo] = (half_t)acc[nt][r];
    }
  }
}

// ---------------------------------------------------------------------------
// Flash attention (causal, online softmax). grid=(TT/128, NB*NH), 256 thr.
// ---------------------------------------------------------------------------
__global__ void __launch_bounds__(256)
k_attn(const half_t* __restrict__ Q, const half_t* __restrict__ Kc,
       const half_t* __restrict__ Vt, half_t* __restrict__ Cc) {
  __shared__ __align__(32) half_t pstage[8][16 * 32];   // per-wave P re-layout
  const int wave = threadIdx.x >> 5;
  const int lane = threadIdx.x & 31;
  const int hi = lane >> 4, nlo = lane & 15;
  const int bh = blockIdx.y;
  const int bidx = bh >> 4, h = bh & 15;
  const int t0 = blockIdx.x * 128 + wave * 16;

  const half_t* q  = Q  + (size_t)bh * TT * HS;
  const half_t* kk = Kc + (size_t)bh * TT * HS;
  const half_t* vt = Vt + (size_t)bh * HS * TT;
  half_t* pbuf = pstage[wave];

  const v16h qa0 = load_a_frag(q + t0 * HS + 0,  HS);
  const v16h qa1 = load_a_frag(q + t0 * HS + 32, HS);

  v8f o[4] = {};
  float rm[8], rl[8];
#pragma unroll
  for (int r = 0; r < 8; ++r) { rm[r] = -1.0e30f; rl[r] = 0.0f; }

  const float scale = 0.125f;   // 1/sqrt(64)
  for (int s0 = 0; s0 <= t0 + 15; s0 += 32) {
    v8f s[2] = {};
#pragma unroll
    for (int j = 0; j < 2; ++j) {
      v16h b0 = load_b_frag(kk + (s0 + j * 16) * HS + 0,  HS);
      v16h b1 = load_b_frag(kk + (s0 + j * 16) * HS + 32, HS);
      s[j] = wmma_f16(qa0, b0, s[j]);
      s[j] = wmma_f16(qa1, b1, s[j]);
    }
    float sc[8];
#pragma unroll
    for (int r = 0; r < 8; ++r) {
      const int trow = t0 + hi * 8 + r;
      float v0 = s[0][r] * scale; if (s0 + nlo      > trow) v0 = -1.0e30f;
      float v1 = s[1][r] * scale; if (s0 + 16 + nlo > trow) v1 = -1.0e30f;
      s[0][r] = v0; s[1][r] = v1;
      float mx = fmaxf(v0, v1);
#pragma unroll
      for (int off = 8; off >= 1; off >>= 1)
        mx = fmaxf(mx, __shfl_xor(mx, off, 32));
      const float nm = fmaxf(rm[r], mx);
      sc[r] = __expf(rm[r] - nm);
      rm[r] = nm;
    }
#pragma unroll
    for (int r = 0; r < 8; ++r) {
      const float p0 = __expf(s[0][r] - rm[r]);
      const float p1 = __expf(s[1][r] - rm[r]);
      float sm = p0 + p1;
#pragma unroll
      for (int off = 8; off >= 1; off >>= 1) sm += __shfl_xor(sm, off, 32);
      rl[r] = rl[r] * sc[r] + sm;
      pbuf[(hi * 8 + r) * 32 + nlo]      = (half_t)p0;
      pbuf[(hi * 8 + r) * 32 + 16 + nlo] = (half_t)p1;
    }
#pragma unroll
    for (int nt = 0; nt < 4; ++nt)
#pragma unroll
      for (int r = 0; r < 8; ++r) o[nt][r] *= sc[r];

    v16h pa = load_a_frag(pbuf, 32);   // wave-private LDS, same-wave ordering
#pragma unroll
    for (int nt = 0; nt < 4; ++nt) {
      v16h bv = load_b_frag(vt + (nt * 16) * TT + s0, TT);
      o[nt] = wmma_f16(pa, bv, o[nt]);
    }
  }

  float inv[8];
#pragma unroll
  for (int r = 0; r < 8; ++r) inv[r] = 1.0f / rl[r];
  half_t* __restrict__ cc = Cc + ((size_t)(bidx * TT + t0)) * EMB + h * HS;
#pragma unroll
  for (int nt = 0; nt < 4; ++nt)
#pragma unroll
    for (int r = 0; r < 8; ++r)
      cc[(hi * 8 + r) * EMB + nt * 16 + nlo] = (half_t)(o[nt][r] * inv[r]);
}

// ---------------------------------------------------------------------------
// Output projection: [BT, 1024] x [1024, 1024] + bias. grid=(BT/256, E/64).
// WoT panel (64 x 1024) staged to LDS via TDM, same as k_qkv.
// ---------------------------------------------------------------------------
__global__ void __launch_bounds__(256)
k_oproj(const half_t* __restrict__ Cc, const half_t* __restrict__ WoT,
        const float* __restrict__ bo, float* __restrict__ out) {
  __shared__ __align__(32) half_t ldsW[64 * LDS_ROW / 2];
  const int wave = threadIdx.x >> 5;
  const int lane = threadIdx.x & 31;
  const int hi = lane >> 4, nlo = lane & 15;
  const int m0 = blockIdx.x * 256 + wave * 32;
  const int n0 = blockIdx.y * 64;

  if (wave == 0) {
    tdm_load_panel(WoT + (size_t)n0 * EMB, lds_addr_of(ldsW), EMB, 64, EMB, 64, EMB);
    __builtin_amdgcn_s_wait_tensorcnt(0);
  }
  __syncthreads();

  v8f acc0[4] = {}, acc1[4] = {};
  gemm_panel2(Cc + (size_t)m0 * EMB, ldsW, acc0, acc1);

#pragma unroll
  for (int mt = 0; mt < 2; ++mt) {
    const v8f* acc = mt ? acc1 : acc0;
#pragma unroll
    for (int nt = 0; nt < 4; ++nt) {
      const float bias = bo[n0 + nt * 16 + nlo];
#pragma unroll
      for (int r = 0; r < 8; ++r)
        out[(size_t)(m0 + mt * 16 + hi * 8 + r) * EMB + n0 + nt * 16 + nlo] =
            acc[nt][r] + bias;
    }
  }
}

// ---------------------------------------------------------------------------
extern "C" void kernel_launch(void* const* d_in, const int* in_sizes, int n_in,
                              void* d_out, int out_size, void* d_ws, size_t ws_size,
                              hipStream_t stream) {
  const float* x  = (const float*)d_in[0];
  const float* Wq = (const float*)d_in[1];
  const float* Wk = (const float*)d_in[2];
  const float* Wv = (const float*)d_in[3];
  const float* Wo = (const float*)d_in[4];
  const float* bo = (const float*)d_in[5];
  float* out = (float*)d_out;

  char* ws = (char*)d_ws;
  size_t off = 0;
  half_t* xh  = (half_t*)(ws + off); off += (size_t)BT * EMB * 2;      // 16 MB (reused as Cc)
  half_t* wtq = (half_t*)(ws + off); off += (size_t)NH * HS * EMB * 2; // 2 MB
  half_t* wtk = (half_t*)(ws + off); off += (size_t)NH * HS * EMB * 2;
  half_t* wtv = (half_t*)(ws + off); off += (size_t)NH * HS * EMB * 2;
  half_t* wot = (half_t*)(ws + off); off += (size_t)EMB * EMB * 2;     // 2 MB
  half_t* Qb  = (half_t*)(ws + off); off += (size_t)NB * NH * TT * HS * 2; // 16 MB
  half_t* Kb  = (half_t*)(ws + off); off += (size_t)NB * NH * TT * HS * 2;
  half_t* Vtb = (half_t*)(ws + off); off += (size_t)NB * NH * TT * HS * 2;
  (void)ws_size; (void)in_sizes; (void)n_in; (void)out_size;

  k_cvt_x<<<dim3(8192), dim3(256), 0, stream>>>(x, xh, BT * EMB);
  k_cvt_w_qkv<<<dim3((NH * EMB * HS + 255) / 256), dim3(256), 0, stream>>>(Wq, wtq);
  k_cvt_w_qkv<<<dim3((NH * EMB * HS + 255) / 256), dim3(256), 0, stream>>>(Wk, wtk);
  k_cvt_w_qkv<<<dim3((NH * EMB * HS + 255) / 256), dim3(256), 0, stream>>>(Wv, wtv);
  k_cvt_wo<<<dim3((EMB * EMB + 255) / 256), dim3(256), 0, stream>>>(Wo, wot);

  k_qkv<<<dim3(BT / 256, NH, 3), dim3(256), 0, stream>>>(xh, wtq, wtk, wtv,
                                                         Qb, Kb, Vtb);
  k_attn<<<dim3(TT / 128, NB * NH), dim3(256), 0, stream>>>(Qb, Kb, Vtb, xh);
  k_oproj<<<dim3(BT / 256, EMB / 64), dim3(256), 0, stream>>>(xh, wot, bo, out);
}
